// GraphATALoss_41042707481216
// MI455X (gfx1250) — matrix operand order repaired and checked by reading, not compiled, not run on hardware
//
#include <hip/hip_runtime.h>
#include <hip/hip_bf16.h>
#include <math.h>

#define BB      8192
#define NHID    512
#define NNODES  50000
#define NCLS    40
#define KTOP    41          // K+1
#define EPSV    1e-5f
#define NEGINF  -3.0e38f
#define CAP     2048

#define TROWS   128                 // B-tile rows (columns of sim) per chunk
#define TPITCH  1040                // 1024B row + 16B TDM pad -> conflict-free ds_load_b128

typedef __attribute__((ext_vector_type(16))) __bf16 v16bf;
typedef __attribute__((ext_vector_type(8)))  float  v8f;
typedef __attribute__((ext_vector_type(4)))  unsigned int v4u;
typedef __attribute__((ext_vector_type(4)))  int v4i;
typedef __attribute__((ext_vector_type(8)))  int v8i;

union Frag16 {
    unsigned int u[8];
    uint4        q[2];
    v16bf        v;
};

__device__ __forceinline__ unsigned short f2bf(float f) {
    unsigned int u = __builtin_bit_cast(unsigned int, f);
    unsigned int r = u + 0x7FFFu + ((u >> 16) & 1u);   // round-to-nearest-even
    return (unsigned short)(r >> 16);
}

// ---- Tensor Data Mover: stream one 128x512 bf16 tile (rows of memn) to LDS.
__device__ __forceinline__ void tdm_load_chunk(const unsigned short* memn,
                                               int chunk, void* ldsbuf) {
    unsigned long long g = (unsigned long long)memn +
                           (unsigned long long)chunk * (unsigned long long)(TROWS * NHID * 2);
    unsigned rem = (unsigned)(NNODES - chunk * TROWS);   // rows left for OOB clamp

    v4u g0;
    g0[0] = 1u;                                          // count=1, user descriptor
    g0[1] = (unsigned)(unsigned long long)ldsbuf;        // LDS byte address (addr[31:0])
    g0[2] = (unsigned)(g & 0xFFFFFFFFull);               // global_addr[31:0]
    g0[3] = (unsigned)((g >> 32) & 0x1FFFFFFull) | (2u << 30);   // addr[56:32] | type=2

    v8i g1;
    g1[0] = (int)((1u << 16) | (1u << 20) | (7u << 22) | (3u << 25));
            // data_size=1(2B) | pad_enable | pad_interval=7(256dw) | pad_amount=3(4dw)
    g1[1] = (int)(512u << 16);                 // tensor_dim0[15:0]  @ bits63:48
    g1[2] = (int)((rem & 0xFFFFu) << 16);      // tensor_dim1[15:0]  @ bits95:80
    g1[3] = (int)(512u << 16);                 // tile_dim0=512      @ bits127:112
    g1[4] = (int)TROWS;                        // tile_dim1=128      @ bits143:128
    g1[5] = (int)512;                          // tensor_dim0_stride @ bits191:160
    g1[6] = 0;
    g1[7] = 0;

    v4i z4 = {0, 0, 0, 0};
#if defined(__clang_major__) && (__clang_major__ >= 23)
    v8i z8 = {0, 0, 0, 0, 0, 0, 0, 0};
    __builtin_amdgcn_tensor_load_to_lds(g0, g1, z4, z4, z8, 0);
#else
    __builtin_amdgcn_tensor_load_to_lds(g0, g1, z4, z4, 0);
#endif
}

// ---------------------------------------------------------------- init
__global__ void init_acc(float* acc) {
    if (threadIdx.x < 64) acc[threadIdx.x] = 0.0f;
}

// ------------------------------------------- entropy + mean-softmax stats
__global__ __launch_bounds__(256) void softmax_stats(const float* __restrict__ cls,
                                                     float* __restrict__ acc) {
    __shared__ float sm[NCLS];
    const int t = threadIdx.x;
    if (t < NCLS) sm[t] = 0.0f;
    __syncthreads();

    const int row = blockIdx.x * blockDim.x + t;
    if (row < BB) {
        const float* x = cls + (size_t)row * NCLS;
        float mx = NEGINF;
        #pragma unroll
        for (int c = 0; c < NCLS; ++c) mx = fmaxf(mx, x[c]);
        float p[NCLS];
        float s = 0.0f;
        #pragma unroll
        for (int c = 0; c < NCLS; ++c) { p[c] = __expf(x[c] - mx); s += p[c]; }
        const float inv = 1.0f / s;
        float ent = 0.0f;
        #pragma unroll
        for (int c = 0; c < NCLS; ++c) {
            float pc = p[c] * inv;
            ent -= pc * __logf(pc + EPSV);
            atomicAdd(&sm[c], pc * (1.0f / BB));
        }
        atomicAdd(&acc[0], ent * (1.0f / BB));
    }
    __syncthreads();
    if (t < NCLS) atomicAdd(&acc[2 + t], sm[t]);
}

// ------------------------------------------------ L2 normalize -> bf16
__global__ __launch_bounds__(256) void l2norm_bf16(const float* __restrict__ in,
                                                   unsigned short* __restrict__ out) {
    __shared__ float red[256];
    const int row = blockIdx.x;
    const float* x = in + (size_t)row * NHID;
    float s = 0.0f;
    for (int i = threadIdx.x; i < NHID; i += 256) { float v = x[i]; s += v * v; }
    red[threadIdx.x] = s;
    __syncthreads();
    for (int w = 128; w > 0; w >>= 1) {
        if (threadIdx.x < w) red[threadIdx.x] += red[threadIdx.x + w];
        __syncthreads();
    }
    const float inv = 1.0f / fmaxf(sqrtf(red[0]), 1e-12f);
    for (int i = threadIdx.x; i < NHID; i += 256)
        out[(size_t)row * NHID + i] = f2bf(x[i] * inv);
}

// --- fused: TDM-staged bf16 WMMA similarity + streaming top-41 + cls loss
__global__ __launch_bounds__(256) void knn_wmma(const unsigned short* __restrict__ featn,
                                                const unsigned short* __restrict__ memn,
                                                const float* __restrict__ cls,
                                                const float* __restrict__ mem_cls,
                                                float* __restrict__ acc) {
    __shared__ alignas(16) unsigned char btile[2][TROWS * TPITCH];  // 2 x 130KB
    __shared__ float topv[16][KTOP];
    __shared__ int   topi[16][KTOP];
    __shared__ int   topc[16];
    __shared__ float thr[16];
    __shared__ float cv[CAP];
    __shared__ int   ck[CAP];
    __shared__ int   ccnt;

    const int tid   = threadIdx.x;
    const int lane  = tid & 31;
    const int wave  = tid >> 5;          // 8 waves
    const int lrow  = lane & 15;
    const int lhalf = lane >> 4;         // 0 or 1
    const int m0    = blockIdx.x * 16;

    if (tid < 16) { topc[tid] = 0; thr[tid] = NEGINF; }
    if (tid == 0) ccnt = 0;

    // ---- A fragments: 16 queries x 512 K, kept in registers, reused 391x
    const int am = m0 + lrow;
    const uint4* a4 = (const uint4*)(featn + (size_t)am * NHID);
    Frag16 afr[16];
    #pragma unroll
    for (int ks = 0; ks < 16; ++ks) {
        const int kb = ks * 32 + lhalf * 16;
        afr[ks].q[0] = a4[kb >> 3];          // K kb   .. kb+7
        afr[ks].q[1] = a4[(kb + 8) >> 3];    // K kb+8 .. kb+15
    }

    const int NCH = (NNODES + TROWS - 1) / TROWS;     // 391 chunks of 128 columns
    if (wave == 0) tdm_load_chunk(memn, 0, btile[0]); // prime the pipeline
    __syncthreads();

    for (int ch = 0; ch < NCH; ++ch) {
        // ---- issue TDM for next chunk into the other buffer, ensure this
        //      chunk's TDM is complete (TENSORcnt retires in order).
        if (wave == 0) {
            if (ch + 1 < NCH) {
                tdm_load_chunk(memn, ch + 1, btile[(ch + 1) & 1]);
                __builtin_amdgcn_s_wait_tensorcnt(1);
            } else {
                __builtin_amdgcn_s_wait_tensorcnt(0);
            }
        }
        if (tid == 0) ccnt = 0;
        __syncthreads();          // tile ready + ccnt reset visible

        // ---- 16x16 WMMA tile per wave; B fragments from LDS with a
        //      3-buffer rotation kept 2 pairs ahead of the consuming WMMA.
        const unsigned char* bbase =
            btile[ch & 1] + (wave * 16 + lrow) * TPITCH + lhalf * 32;

        Frag16 bf[3];
        #pragma unroll
        for (int p = 0; p < 3; ++p) {
            bf[p].q[0] = *(const uint4*)(bbase + p * 64);
            bf[p].q[1] = *(const uint4*)(bbase + p * 64 + 16);
        }

        v8f c = {0.f, 0.f, 0.f, 0.f, 0.f, 0.f, 0.f, 0.f};
        #pragma unroll
        for (int ks = 0; ks < 16; ++ks) {
            c = __builtin_amdgcn_wmma_f32_16x16x32_bf16(
                    false, afr[ks].v, false, bf[ks % 3].v, (short)0, c, false, false);
            if (ks + 3 < 16) {
                bf[(ks + 3) % 3].q[0] = *(const uint4*)(bbase + (ks + 3) * 64);
                bf[(ks + 3) % 3].q[1] = *(const uint4*)(bbase + (ks + 3) * 64 + 16);
            }
        }
        // Pin the schedule: 6 DS reads up front, then {1 WMMA, 2 DS}x13,
        // then the 3 tail WMMAs -> waits relax to s_wait_dscnt <= 4.
        __builtin_amdgcn_sched_group_barrier(0x100, 6, 0);   // DS read x6
        #pragma unroll
        for (int i = 0; i < 13; ++i) {
            __builtin_amdgcn_sched_group_barrier(0x008, 1, 0);   // WMMA
            __builtin_amdgcn_sched_group_barrier(0x100, 2, 0);   // DS read x2
        }
        __builtin_amdgcn_sched_group_barrier(0x008, 1, 0);
        __builtin_amdgcn_sched_group_barrier(0x008, 1, 0);
        __builtin_amdgcn_sched_group_barrier(0x008, 1, 0);

        // ---- threshold filter + candidate append
        const int nblock = ch * TROWS + wave * 16;
        const int nn     = nblock + lrow;             // this lane's column
        #pragma unroll
        for (int r = 0; r < 8; ++r) {
            const int mrow = r + lhalf * 8;           // C layout: VGPR r -> M
            const float v = c[r];
            if (nn < NNODES && v > thr[mrow]) {
                int slot = atomicAdd(&ccnt, 1);
                if (slot < CAP) { cv[slot] = v; ck[slot] = (mrow << 20) | nn; }
            }
        }
        __syncthreads();

        // ---- one owner thread per query row merges its candidates
        if (tid < 16) {
            int cnt = ccnt; if (cnt > CAP) cnt = CAP;
            int   tc   = topc[tid];
            float tmin = thr[tid];
            for (int i = 0; i < cnt; ++i) {
                const int k = ck[i];
                if ((k >> 20) != tid) continue;
                const float v = cv[i];
                const int  n = k & 0xFFFFF;
                if (tc < KTOP) {
                    topv[tid][tc] = v; topi[tid][tc] = n; ++tc;
                    if (tc == KTOP) {
                        tmin = 3e38f;
                        for (int j = 0; j < KTOP; ++j) tmin = fminf(tmin, topv[tid][j]);
                    }
                } else if (v > tmin) {
                    int ms = 0; float mv = topv[tid][0];
                    for (int j = 1; j < KTOP; ++j)
                        if (topv[tid][j] < mv) { mv = topv[tid][j]; ms = j; }
                    topv[tid][ms] = v; topi[tid][ms] = n;
                    tmin = 3e38f;
                    for (int j = 0; j < KTOP; ++j) tmin = fminf(tmin, topv[tid][j]);
                }
            }
            topc[tid] = tc;
            thr[tid]  = (tc == KTOP) ? tmin : NEGINF;
        }
        __syncthreads();          // owners done before next chunk's reset/append
    }

    // ---- epilogue: drop nearest, average neighbor classes, cls loss
    if (tid < 16) {
        const int row = m0 + tid;
        int ms = 0; float mv = topv[tid][0];
        for (int j = 1; j < KTOP; ++j)
            if (topv[tid][j] > mv) { mv = topv[tid][j]; ms = j; }
        float s[NCLS];
        #pragma unroll
        for (int cc = 0; cc < NCLS; ++cc) s[cc] = 0.0f;
        for (int j = 0; j < KTOP; ++j) {
            if (j == ms) continue;
            const float* mc = mem_cls + (size_t)topi[tid][j] * NCLS;
            #pragma unroll
            for (int cc = 0; cc < NCLS; ++cc) s[cc] += mc[cc];
        }
        int pred = 0; float pv = s[0];
        for (int cc = 1; cc < NCLS; ++cc)
            if (s[cc] > pv) { pv = s[cc]; pred = cc; }

        const float* x = cls + (size_t)row * NCLS;
        float mx = NEGINF;
        for (int cc = 0; cc < NCLS; ++cc) mx = fmaxf(mx, x[cc]);
        float se = 0.0f;
        for (int cc = 0; cc < NCLS; ++cc) se += __expf(x[cc] - mx);
        const float lse = mx + __logf(se);
        atomicAdd(&acc[1], -(x[pred] - lse) * (1.0f / BB));
    }
}

// ---------------------------------------------------------------- final
__global__ void finalize(const float* __restrict__ acc, float* __restrict__ out) {
    if (threadIdx.x == 0 && blockIdx.x == 0) {
        float div = 0.0f;
        for (int c = 0; c < NCLS; ++c) {
            const float m = acc[2 + c];
            div += m * __logf(m + EPSV);
        }
        out[0] = acc[0] + div + acc[1];   // entropy + div + cls
    }
}

extern "C" void kernel_launch(void* const* d_in, const int* in_sizes, int n_in,
                              void* d_out, int out_size, void* d_ws, size_t ws_size,
                              hipStream_t stream) {
    const float* feat  = (const float*)d_in[0];   // [8192,512]
    const float* cls   = (const float*)d_in[1];   // [8192,40]
    const float* memf  = (const float*)d_in[2];   // [50000,512]
    const float* memc  = (const float*)d_in[3];   // [50000,40]
    float* out = (float*)d_out;

    float* acc = (float*)d_ws;                                        // 64 floats
    unsigned short* featn = (unsigned short*)((char*)d_ws + 256);     // 8 MB
    unsigned short* memn  = featn + (size_t)BB * NHID;                // 51.2 MB

    init_acc<<<1, 64, 0, stream>>>(acc);
    softmax_stats<<<BB / 256, 256, 0, stream>>>(cls, acc);
    l2norm_bf16<<<BB,     256, 0, stream>>>(feat, featn);
    l2norm_bf16<<<NNODES, 256, 0, stream>>>(memf, memn);
    knn_wmma<<<BB / 16, 256, 0, stream>>>(featn, memn, cls, memc, acc);
    finalize<<<1, 1, 0, stream>>>(acc, out);
}
